// GasAbsorption_27977416966531
// MI455X (gfx1250) — compile-verified, matrix operand order checked by reading, not run on hardware
//
#include <hip/hip_runtime.h>
#include <hip/hip_bf16.h>
#include <math.h>

#define N_WL   262144
#define T_REF  273.15f
#define P_REF  101325.0f
#define C_LIGHT 299792458.0f

typedef __attribute__((ext_vector_type(16))) _Float16 v16h;
typedef __attribute__((ext_vector_type(8)))  float    v8f;

// Fast hardware reciprocal (v_rcp_f32, ~1 ulp) instead of the ~10-op IEEE
// division expansion. All divides here feed smooth nonlinearities and are
// already below the fp16 WMMA quantization noise.
__device__ __forceinline__ float frcp_(float x) { return __builtin_amdgcn_rcpf(x); }

// ---------------------------------------------------------------------------
// ws layout (floats):
//   [0..31]    h            (continuum hidden vector, post-silu)
//   [32..95]   m2           (mixing hidden vector, post-silu)
//   [96..105]  cf[l]  = conc*strength_T / (sigma*sqrt(pi)+1e-12)
//   [106..115] nu0[l]
//   [116..125] isg[l] = 1/(sigma+1e-12)
//   [126..135] yv[l]  = gamma_L/(sigma+1e-12)
//   [146..153] cross8 (cross incl. continuum at first 8 wavelengths)
//   [154..217] m1 scratch
// ---------------------------------------------------------------------------

__device__ __forceinline__ float sigmoidf_(float x) { return frcp_(1.0f + __expf(-x)); }
__device__ __forceinline__ float siluf_(float x)    { return x * sigmoidf_(x); }
// Branchless, overflow-safe softplus: max(x,0) + log(1 + exp(-|x|)).
__device__ __forceinline__ float softplusf_(float x) {
    return fmaxf(x, 0.0f) + __logf(1.0f + __expf(-fabsf(x)));
}

// Voigt (Humlicek-style 3-region approximation from the reference), summed
// over the 10 lines with the combined coefficient folded in.
__device__ __forceinline__ float voigt_sum(float wl, const float* cf, const float* nu0,
                                           const float* isg, const float* yv) {
    float acc = 0.0f;
#pragma unroll
    for (int l = 0; l < 10; ++l) {
        float x  = (wl - nu0[l]) * isg[l];
        float y  = yv[l];
        float ax = fabsf(x);
        float s  = ax + y;
        // t = y - i x ;  u = t^2
        float ur = y * y - x * x;
        float ui = -2.0f * x * y;
        // region 1: w = Re( t*c / (0.5 + t^2) ),  c = 0.5641896
        float dr   = 0.5f + ur;
        float di   = ui;
        float inv1 = frcp_(dr * dr + di * di);
        float w1   = (0.5641896f * y * dr - 0.5641896f * x * di) * inv1;
        // region 2: w = Re( t*(1.410474 + 0.5641896 u) / (0.75 + u(3+u)) )
        float pr  = 1.410474f + 0.5641896f * ur;
        float pim = 0.5641896f * ui;
        float n2r = y * pr + x * pim;
        float n2i = y * pim - x * pr;
        float u2r = ur * ur - ui * ui;
        float u2i = 2.0f * ur * ui;
        float d2r = 0.75f + 3.0f * ur + u2r;
        float d2i = 3.0f * ui + u2i;
        float w2  = (n2r * d2r + n2i * d2i) * frcp_(d2r * d2r + d2i * d2i);
        // region 3 (near line center)
        float x2 = x * x;
        float w3 = __expf(-x2) * __cosf(2.0f * x * y) * 0.56418958f
                 + 0.63661977f * y * __sinf(x2) * frcp_(x2 + y * y + 1e-10f);
        float w = (s >= 15.0f) ? w1 : ((ax >= 5.5f) ? w2 : w3);
        acc += cf[l] * w;
    }
    return acc;
}

// ---------------------------------------------------------------------------
// Kernel 1: tiny sequential prelude (line params, h, cross[:8], m1, m2).
// One block of 64 threads; cost is negligible vs. the streaming pass.
// ---------------------------------------------------------------------------
__global__ void gas_prep_kernel(const float* __restrict__ wavelengths,
                                const float* __restrict__ Tp, const float* __restrict__ Pp,
                                const float* __restrict__ o3p, const float* __restrict__ h2op,
                                const float* __restrict__ co2p,
                                const float* __restrict__ mix_w1, const float* __restrict__ mix_b1,
                                const float* __restrict__ mix_w2, const float* __restrict__ mix_b2,
                                const float* __restrict__ cont_w1, const float* __restrict__ cont_b1,
                                const float* __restrict__ cont_w2, const float* __restrict__ cont_b2,
                                float* __restrict__ ws) {
    const int t = threadIdx.x;
    const float T = *Tp, P = *Pp;
    const float c_h2o = *h2op;

    // ---- stage A: per-line parameters -------------------------------------
    if (t < 10) {
        const float LN[10][4] = {
            {254.0f, 1.15e-17f, 2.0f, 0.05f}, {280.0f, 5e-18f,   3.0f, 0.04f},
            {310.0f, 1.9e-19f,  2.5f, 0.03f}, {940.0f, 2.5e-23f, 3.0f, 0.40f},
            {1130.0f, 8.2e-24f, 2.5f, 0.35f}, {1380.0f, 1.8e-22f, 4.0f, 0.45f},
            {1400.0f, 3.5e-25f, 3.0f, 0.50f}, {1600.0f, 7.8e-26f, 2.5f, 0.48f},
            {2000.0f, 4.2e-24f, 4.0f, 0.52f}, {2700.0f, 1.2e-24f, 3.5f, 0.49f}};
        const float MASSV[10] = {48.f, 48.f, 48.f, 18.f, 18.f, 18.f, 44.f, 44.f, 44.f, 44.f};
        float conc = (t < 3) ? *o3p : ((t < 6) ? c_h2o : *co2p);
        float nu0 = LN[t][0], S = LN[t][1], width = LN[t][2], texp = LN[t][3];
        float sT = S * powf(T_REF / (T + 1e-12f), texp);
        float gL = width * (P / (P_REF + 1e-12f)) * sqrtf(T_REF / (T + 1e-12f));
        // 2*k_B*N_A = 16.62905...
        float gD = nu0 / C_LIGHT * sqrtf(16.629053f * T / (MASSV[t] + 1e-12f));
        float sigma = gD / (1.1774100f + 1e-12f);   // sqrt(2 ln 2)
        ws[96 + t]  = conc * sT / (sigma * 1.7724539f + 1e-12f); // fold 1/(sigma*sqrt(pi))
        ws[106 + t] = nu0;
        ws[116 + t] = 1.0f / (sigma + 1e-12f);
        ws[126 + t] = gL / (sigma + 1e-12f);
    }
    __syncthreads();

    // ---- stage B: h = silu(cont_feat @ cont_w1 + b1) ----------------------
    if (t < 32) {
        float cfeat[5] = {T / (T_REF + 1e-12f), P / (P_REF + 1e-12f), c_h2o, 1.0f, 0.0f};
        float acc = cont_b1[t];
#pragma unroll
        for (int i = 0; i < 5; ++i) acc += cfeat[i] * cont_w1[i * 32 + t];
        ws[t] = siluf_(acc);
    }
    __syncthreads();

    // ---- stage C: cross[:8] (voigt + continuum at first 8 wavelengths) ----
    if (t < 8) {
        float wl = wavelengths[t];
        float vs = voigt_sum(wl, ws + 96, ws + 106, ws + 116, ws + 126);
        float cd = cont_b2[t];
        for (int k = 0; k < 32; ++k) cd += ws[k] * cont_w2[(size_t)k * N_WL + t];
        ws[146 + t] = vs + softplusf_(cd);
    }
    __syncthreads();

    // ---- stage D: m1 = silu(mix_feat @ mix_w1 + b1) -----------------------
    if (t < 64) {
        float mf[10];
        mf[0] = T / (T_REF + 1e-12f);
        mf[1] = P / (P_REF + 1e-12f);
#pragma unroll
        for (int j = 0; j < 8; ++j) mf[2 + j] = ws[146 + j];
        float acc = mix_b1[t];
#pragma unroll
        for (int i = 0; i < 10; ++i) acc += mf[i] * mix_w1[i * 64 + t];
        ws[154 + t] = siluf_(acc);
    }
    __syncthreads();

    // ---- stage E: m2 = silu(m1 @ mix_w2 + b2) -----------------------------
    if (t < 64) {
        float acc = mix_b2[t];
        for (int k = 0; k < 64; ++k) acc += ws[154 + k] * mix_w2[k * 64 + t];
        ws[32 + t] = siluf_(acc);
    }
}

// ---------------------------------------------------------------------------
// Kernel 2: the streaming pass. Each wave handles 32 wavelengths per chunk:
//  - two 16-col WMMA tiles for mixing (K=64 -> 2 chained 16x16x32 WMMAs each)
//  - two 16-col WMMA tiles for continuum (K=32 -> 1 WMMA each)
//  - per-lane 10-line Voigt evaluation for its own wavelength
// B operand = weight tile (row-major rows -> "row striped across lanes",
// coalesced along wavelength). A operand = broadcast of m/h across the 16
// rows (loop-invariant). Result column n == dot for wavelength wlb+n,
// replicated down rows, so lane L reads element 0 of the matching tile.
// ---------------------------------------------------------------------------

// VGPR<->K mapping for 16-bit 16x32 operands (half-pair v, lane group g):
// v in 0..3 -> K = 2v (+g), v in 4..7 -> K = 16+2(v-4) (+g); g = 0 or 8.
__device__ __forceinline__ int kmap(int v, int g) {
    return ((v < 4) ? (2 * v) : (16 + 2 * (v - 4))) + g;
}

__device__ __forceinline__ v16h broadcast_a(const float* vec, int k0, int g) {
    v16h a;
#pragma unroll
    for (int v = 0; v < 8; ++v) {
        int k = k0 + kmap(v, g);
        a[2 * v]     = (_Float16)vec[k];
        a[2 * v + 1] = (_Float16)vec[k + 1];
    }
    return a;
}

__device__ __forceinline__ v16h load_b_tile(const float* __restrict__ W, int k0,
                                            size_t col, int g) {
    v16h b;
#pragma unroll
    for (int v = 0; v < 8; ++v) {
        int r = k0 + kmap(v, g);
        b[2 * v]     = (_Float16)W[(size_t)r * N_WL + col];
        b[2 * v + 1] = (_Float16)W[(size_t)(r + 1) * N_WL + col];
    }
    return b;
}

__global__ __launch_bounds__(256) void gas_spectrum_kernel(
        const float* __restrict__ wavelengths,
        const float* __restrict__ mix_w3, const float* __restrict__ mix_b3,
        const float* __restrict__ cont_w2, const float* __restrict__ cont_b2,
        const float* __restrict__ ws, float* __restrict__ out) {
    const int lane = threadIdx.x & 31;
    const int g    = (lane & 16) ? 8 : 0;

    // Loop-invariant Voigt parameters into registers.
    float cf[10], nu0[10], isg[10], yv[10];
#pragma unroll
    for (int l = 0; l < 10; ++l) {
        cf[l]  = ws[96 + l];
        nu0[l] = ws[106 + l];
        isg[l] = ws[116 + l];
        yv[l]  = ws[126 + l];
    }
    // Loop-invariant broadcast A operands (h for continuum, m2 for mixing).
    const v16h a_c  = broadcast_a(ws + 0, 0, g);    // K = 0..31  of h
    const v16h a_m0 = broadcast_a(ws + 32, 0, g);   // K = 0..31  of m2
    const v16h a_m1 = broadcast_a(ws + 32, 32, g);  // K = 32..63 of m2

    const int wid = (int)((blockIdx.x * blockDim.x + threadIdx.x) >> 5);
    const int nw  = (int)((gridDim.x * blockDim.x) >> 5);
    const int nchunks = N_WL / 32;

    for (int chunk = wid; chunk < nchunks; chunk += nw) {
        const int    wlb  = chunk * 32;
        const size_t col0 = (size_t)wlb + (lane & 15);
        const size_t col1 = col0 + 16;

        // mixing dot: tile0 (cols wlb..wlb+15), tile1 (cols wlb+16..wlb+31)
        v8f cm0 = {};
        cm0 = __builtin_amdgcn_wmma_f32_16x16x32_f16(false, a_m0, false,
                  load_b_tile(mix_w3, 0, col0, g), (short)0, cm0, false, false);
        cm0 = __builtin_amdgcn_wmma_f32_16x16x32_f16(false, a_m1, false,
                  load_b_tile(mix_w3, 32, col0, g), (short)0, cm0, false, false);
        v8f cm1 = {};
        cm1 = __builtin_amdgcn_wmma_f32_16x16x32_f16(false, a_m0, false,
                  load_b_tile(mix_w3, 0, col1, g), (short)0, cm1, false, false);
        cm1 = __builtin_amdgcn_wmma_f32_16x16x32_f16(false, a_m1, false,
                  load_b_tile(mix_w3, 32, col1, g), (short)0, cm1, false, false);

        // continuum dot (K = 32)
        v8f cc0 = {};
        cc0 = __builtin_amdgcn_wmma_f32_16x16x32_f16(false, a_c, false,
                  load_b_tile(cont_w2, 0, col0, g), (short)0, cc0, false, false);
        v8f cc1 = {};
        cc1 = __builtin_amdgcn_wmma_f32_16x16x32_f16(false, a_c, false,
                  load_b_tile(cont_w2, 0, col1, g), (short)0, cc1, false, false);

        // Lane L owns wavelength wlb+L.
        // D element 0: lanes 0-15 -> row 0 col L (tile0); lanes 16-31 -> row 8
        // col L-16 (tile1). All rows of D are identical (A rows broadcast).
        const float mixdot  = (lane < 16) ? cm0[0] : cm1[0];
        const float contdot = (lane < 16) ? cc0[0] : cc1[0];

        const int   wli = wlb + lane;
        const float wl  = wavelengths[wli];
        const float vs  = voigt_sum(wl, cf, nu0, isg, yv);
        const float cross  = vs + softplusf_(contdot + cont_b2[wli]);
        const float mixing = sigmoidf_(mixdot + mix_b3[wli]);
        out[wli] = cross * (1.0f + 0.1f * (mixing - 0.5f));
    }
}

// ---------------------------------------------------------------------------
extern "C" void kernel_launch(void* const* d_in, const int* in_sizes, int n_in,
                              void* d_out, int out_size, void* d_ws, size_t ws_size,
                              hipStream_t stream) {
    const float* wavelengths = (const float*)d_in[0];
    const float* temperature = (const float*)d_in[1];
    const float* pressure    = (const float*)d_in[2];
    const float* conc_o3     = (const float*)d_in[3];
    const float* conc_h2o    = (const float*)d_in[4];
    const float* conc_co2    = (const float*)d_in[5];
    const float* mix_w1      = (const float*)d_in[6];
    const float* mix_b1      = (const float*)d_in[7];
    const float* mix_w2      = (const float*)d_in[8];
    const float* mix_b2      = (const float*)d_in[9];
    const float* mix_w3      = (const float*)d_in[10];
    const float* mix_b3      = (const float*)d_in[11];
    const float* cont_w1     = (const float*)d_in[12];
    const float* cont_b1     = (const float*)d_in[13];
    const float* cont_w2     = (const float*)d_in[14];
    const float* cont_b2     = (const float*)d_in[15];
    float* out = (float*)d_out;
    float* ws  = (float*)d_ws;

    gas_prep_kernel<<<1, 64, 0, stream>>>(wavelengths, temperature, pressure,
                                          conc_o3, conc_h2o, conc_co2,
                                          mix_w1, mix_b1, mix_w2, mix_b2,
                                          cont_w1, cont_b1, cont_w2, cont_b2, ws);

    // 8192 waves == 8192 chunks of 32 wavelengths: one chunk per wave.
    gas_spectrum_kernel<<<1024, 256, 0, stream>>>(wavelengths, mix_w3, mix_b3,
                                                  cont_w2, cont_b2, ws, out);
}
